// CausalSelfAttention_64338610094596
// MI455X (gfx1250) — matrix operand order repaired and checked by reading, not compiled, not run on hardware
//
#include <hip/hip_runtime.h>

// FoX-style causal attention for MI455X (gfx1250), wave32 + WMMA + async-LDS.
// B=2, T=2048, C=1024, H=16, D=64.
//
// bias_ij = S_i - P_ij with P_ij the causal inclusive prefix sum of
// slope*logsigmoid(p_i . p_rot_j / D) along j; S_i is per-row constant and
// cancels in softmax => flash-attention single pass with a running prefix.
//
// Attention j-tiles of K and P_rot are staged into per-wave LDS with
// global_load_async_to_lds_b128 (double buffered, gated by s_wait_asynccnt)
// when the toolchain exposes the builtin; otherwise synchronous LDS staging.

#define TT 2048
#define CC 1024
#define HH 16
#define DD 64

typedef _Float16 v16h __attribute__((ext_vector_type(16)));
typedef _Float16 v8h  __attribute__((ext_vector_type(8)));
typedef float    v8f  __attribute__((ext_vector_type(8)));
typedef int      v4i  __attribute__((ext_vector_type(4)));

#if defined(__HIP_DEVICE_COMPILE__) && \
    __has_builtin(__builtin_amdgcn_global_load_async_to_lds_b128)
#define FOX_ASYNC 1
#else
#define FOX_ASYNC 0
#endif

static __device__ inline v16h cat8(v8h a, v8h b) {
  return __builtin_shufflevector(a, b, 0,1,2,3,4,5,6,7,8,9,10,11,12,13,14,15);
}
static __device__ inline v16h ld16(const _Float16* p) { return *(const v16h*)p; }
static __device__ inline v8h  ld8 (const _Float16* p) { return *(const v8h*)p; }

static __device__ inline v8f wmma32(v16h a, v16h b, v8f c) {
  // emits v_wmma_f32_16x16x32_f16
  return __builtin_amdgcn_wmma_f32_16x16x32_f16(false, a, false, b, (short)0, c,
                                                false, false);
}

// 16 bytes per lane, global -> LDS (async when available).
static __device__ inline void acp16(const _Float16* g, _Float16* l) {
#if FOX_ASYNC
  __builtin_amdgcn_global_load_async_to_lds_b128(
      (__attribute__((address_space(1))) v4i*)(v4i*)g,
      (__attribute__((address_space(3))) v4i*)(v4i*)l, 0, 0);
#else
  *(v8h*)l = *(const v8h*)g;
#endif
}

template <int N> static __device__ inline void wait_async() {
#if FOX_ASYNC
#if __has_builtin(__builtin_amdgcn_s_wait_asynccnt)
  __builtin_amdgcn_s_wait_asynccnt(N);
#else
  asm volatile("s_wait_asynccnt %0" ::"i"(N) : "memory");
#endif
#endif
}

static __device__ inline float scan16(float x, int l) {
#pragma unroll
  for (int off = 1; off < 16; off <<= 1) {
    float y = __shfl_up(x, off, 16);
    if (l >= off) x += y;
  }
  return x;
}
static __device__ inline float rmax16(float x) {
#pragma unroll
  for (int off = 8; off >= 1; off >>= 1) x = fmaxf(x, __shfl_xor(x, off, 16));
  return x;
}
static __device__ inline float rsum16(float x) {
#pragma unroll
  for (int off = 8; off >= 1; off >>= 1) x += __shfl_xor(x, off, 16);
  return x;
}

// ---------------------------------------------------------------- converts
__global__ void fox_f32_to_f16(const float* __restrict__ in,
                               _Float16* __restrict__ out, int n) {
  int i = (blockIdx.x * blockDim.x + threadIdx.x) * 4;
  if (i + 3 < n) {
    float4 v = *(const float4*)(in + i);
    out[i + 0] = (_Float16)v.x;
    out[i + 1] = (_Float16)v.y;
    out[i + 2] = (_Float16)v.z;
    out[i + 3] = (_Float16)v.w;
  } else {
    for (int k = i; k < n; ++k) out[k] = (_Float16)in[k];
  }
}

// ------------------------------------------------- fused projection GEMM
// Wave: 16 rows x 64 cols (one full head along N), K-loop over 1024.
// Epilogue fuses RMS norm (q,k), raw-p + RMS + rotary (p), transposed v.
__global__ __launch_bounds__(256) void fox_proj_kernel(
    const _Float16* __restrict__ xh,
    const _Float16* __restrict__ whq, const _Float16* __restrict__ whk,
    const _Float16* __restrict__ whv, const _Float16* __restrict__ whp,
    const float* __restrict__ qw, const float* __restrict__ kw,
    _Float16* __restrict__ qh, _Float16* __restrict__ kh,
    _Float16* __restrict__ vt, _Float16* __restrict__ ph,
    _Float16* __restrict__ prh) {
  const int lane = threadIdx.x & 31;
  const int wave = threadIdx.x >> 5;
  const int n16  = lane & 15;
  const int hi   = lane >> 4;
  const int head = blockIdx.x;              // 0..15
  const int mt   = blockIdx.y * 8 + wave;   // 0..255
  const int z    = blockIdx.z;              // 0..3 : q,k,v,p
  const int i0   = mt * 16;
  const int nbase = head * DD;

  const _Float16* W = (z == 0) ? whq : (z == 1) ? whk : (z == 2) ? whv : whp;

  v8f acc[4];
#pragma unroll
  for (int t = 0; t < 4; ++t) acc[t] = v8f{};

  const _Float16* arow = xh + (size_t)(i0 + n16) * CC;
  for (int kk = 0; kk < CC; kk += 32) {
    __builtin_prefetch(arow + kk + 128, 0, 1);  // global_prefetch_b8
    v16h a = cat8(ld8(arow + kk + hi * 16), ld8(arow + kk + hi * 16 + 8));
#pragma unroll
    for (int t = 0; t < 4; ++t) {
      v16h b = ld16(W + (size_t)(nbase + 16 * t + n16) * CC + kk + hi * 16);
      acc[t] = wmma32(a, b, acc[t]);
    }
  }

  float ssq[8];
#pragma unroll
  for (int v = 0; v < 8; ++v) {
    float s = 0.f;
#pragma unroll
    for (int t = 0; t < 4; ++t) { float x = acc[t][v]; s += x * x; }
    ssq[v] = rsum16(s);
  }

  if (z == 0 || z == 1) {
    const float* rw = (z == 0) ? qw : kw;
    float w4[4];
#pragma unroll
    for (int t = 0; t < 4; ++t) w4[t] = rw[16 * t + n16];
    _Float16* dst = (z == 0) ? qh : kh;
#pragma unroll
    for (int v = 0; v < 8; ++v) {
      int it = i0 + v + 8 * hi;
      int b = it >> 11, tt = it & (TT - 1);
      size_t rb = ((size_t)(b * HH + head) * TT + tt) * DD;
      float sc = rsqrtf(ssq[v] * (1.f / 64.f) + 1e-5f);
#pragma unroll
      for (int t = 0; t < 4; ++t)
        dst[rb + 16 * t + n16] = (_Float16)(acc[t][v] * sc * w4[t]);
    }
  } else if (z == 2) {
#pragma unroll
    for (int v = 0; v < 8; ++v) {
      int it = i0 + v + 8 * hi;
      int b = it >> 11, tt = it & (TT - 1);
      size_t pb = (size_t)(b * HH + head) * DD * TT;
#pragma unroll
      for (int t = 0; t < 4; ++t)
        vt[pb + (size_t)(16 * t + n16) * TT + tt] = (_Float16)acc[t][v];
    }
  } else {
#pragma unroll
    for (int v = 0; v < 8; ++v) {
      int it = i0 + v + 8 * hi;
      int b = it >> 11, tt = it & (TT - 1);
      size_t rb = ((size_t)(b * HH + head) * TT + tt) * DD;
      float sc = rsqrtf(ssq[v] * (1.f / 64.f) + 1.1920929e-7f);
      float c = cosf((float)tt), s = sinf((float)tt);
      float pn[4];
#pragma unroll
      for (int t = 0; t < 4; ++t) {
        ph[rb + 16 * t + n16] = (_Float16)acc[t][v];  // raw p for the A side
        pn[t] = acc[t][v] * sc;
      }
      prh[rb + 0 * 16 + n16] = (_Float16)(pn[0] * c + pn[2] * s);
      prh[rb + 1 * 16 + n16] = (_Float16)(pn[1] * c + pn[3] * s);
      prh[rb + 2 * 16 + n16] = (_Float16)(pn[2] * c - pn[0] * s);
      prh[rb + 3 * 16 + n16] = (_Float16)(pn[3] * c - pn[1] * s);
    }
  }
}

// ------------------------------------------------------- flash attention
// One wave per 16-row tile of one (b,head); j advances in 32-col steps.
// K / P_rot j-tiles (4KB each, contiguous in [B,H,T,D]) are async-staged
// into per-wave LDS, double buffered: 16 b128 transfers per stage, so
// s_wait_asynccnt 16 completes the current tile while the next is in flight.
__global__ __launch_bounds__(128) void fox_attn_kernel(
    const _Float16* __restrict__ qh, const _Float16* __restrict__ kh,
    const _Float16* __restrict__ ph, const _Float16* __restrict__ prh,
    const _Float16* __restrict__ vt, _Float16* __restrict__ yh) {
  __shared__ __align__(32) _Float16 kst[4][2][32 * DD];  // 32KB
  __shared__ __align__(32) _Float16 pst[4][2][32 * DD];  // 32KB
  __shared__ __align__(32) _Float16 pbuf[4][16 * 32];    // 4KB
  const int lane = threadIdx.x & 31;
  const int wave = threadIdx.x >> 5;
  const int n16  = lane & 15;
  const int hi   = lane >> 4;
  const int head = blockIdx.y;
  const int b    = blockIdx.z;
  const int rt   = blockIdx.x * 4 + wave;  // 0..127
  const int i0   = rt * 16;

  const size_t plane  = (size_t)(b * HH + head) * TT * DD;
  const size_t vplane = (size_t)(b * HH + head) * DD * TT;
  const float slope = exp2f(-0.5f * (float)(head + 1));  // ALiBi, H=16

  const _Float16* qrow = qh + plane + (size_t)(i0 + n16) * DD;
  const _Float16* prow = ph + plane + (size_t)(i0 + n16) * DD;
  v16h aq[2], apm[2];
#pragma unroll
  for (int kkk = 0; kkk < 2; ++kkk) {
    aq[kkk]  = cat8(ld8(qrow + kkk * 32 + hi * 16), ld8(qrow + kkk * 32 + hi * 16 + 8));
    apm[kkk] = cat8(ld8(prow + kkk * 32 + hi * 16), ld8(prow + kkk * 32 + hi * 16 + 8));
  }

  v8f acc[4];
#pragma unroll
  for (int t = 0; t < 4; ++t) acc[t] = v8f{};
  float mrow[8], lrow[8], carry[8];
#pragma unroll
  for (int v = 0; v < 8; ++v) { mrow[v] = -1e30f; lrow[v] = 0.f; carry[v] = 0.f; }

  // stage j-tile `j0` into buffer `buf` (16 async b128 ops)
  auto stage = [&](int j0, int buf) {
    const _Float16* gk = kh  + plane + (size_t)j0 * DD;
    const _Float16* gp = prh + plane + (size_t)j0 * DD;
    _Float16* lk = &kst[wave][buf][0];
    _Float16* lp = &pst[wave][buf][0];
#pragma unroll
    for (int u = 0; u < 8; ++u) {
      int e = (u * 32 + lane) * 8;  // halves; 16B per lane
      acp16(gk + e, lk + e);
      acp16(gp + e, lp + e);
    }
  };

  const int ilast = i0 + 15;
  const int nsteps = ilast / 32 + 1;
  stage(0, 0);

  for (int step = 0; step < nsteps; ++step) {
    const int j0 = step * 32;
    const int buf = step & 1;
    if (step + 1 < nsteps) {
      stage(j0 + 32, buf ^ 1);
      wait_async<16>();  // current tile complete; next 16 still in flight
    } else {
      wait_async<0>();
    }
    const _Float16* lk = &kst[wave][buf][0];
    const _Float16* lp = &pst[wave][buf][0];

    v8f sq[2], sp[2];
#pragma unroll
    for (int s = 0; s < 2; ++s) {
      sq[s] = v8f{}; sp[s] = v8f{};
#pragma unroll
      for (int kkk = 0; kkk < 2; ++kkk) {
        int o = (16 * s + n16) * DD + kkk * 32 + hi * 16;
        sq[s] = wmma32(aq[kkk],  ld16(lk + o), sq[s]);
        sp[s] = wmma32(apm[kkk], ld16(lp + o), sp[s]);
      }
    }

#pragma unroll
    for (int v = 0; v < 8; ++v) {
      const int gi = i0 + v + 8 * hi;
      float fr[2];
#pragma unroll
      for (int s = 0; s < 2; ++s) {
        int gj = j0 + 16 * s + n16;
        float x = sp[s][v] * (1.f / 64.f);
        float ls = fminf(x, 0.f) - __logf(1.f + __expf(-fabsf(x)));
        float f = slope * ls;
        if (gj > gi) f = 0.f;
        fr[s] = f;
      }
      float P0 = scan16(fr[0], n16);
      float tot0 = __shfl(P0, 15, 16);
      float P1 = scan16(fr[1], n16) + tot0;
      float tot1 = __shfl(P1, 15, 16);
      float zz0 = sq[0][v] * 0.125f - (carry[v] + P0);
      float zz1 = sq[1][v] * 0.125f - (carry[v] + P1);
      carry[v] += tot1;
      if (j0 + n16      > gi) zz0 = -1e30f;
      if (j0 + 16 + n16 > gi) zz1 = -1e30f;

      float mn = fmaxf(mrow[v], rmax16(fmaxf(zz0, zz1)));
      float scale = __expf(mrow[v] - mn);
      float p0 = __expf(zz0 - mn);
      float p1 = __expf(zz1 - mn);
      lrow[v] = lrow[v] * scale + rsum16(p0 + p1);
      mrow[v] = mn;
#pragma unroll
      for (int t = 0; t < 4; ++t) acc[t][v] *= scale;

      int m = v + 8 * hi;
      pbuf[wave][m * 32 + n16]      = (_Float16)p0;
      pbuf[wave][m * 32 + 16 + n16] = (_Float16)p1;
    }

    // probs: C-layout -> A-layout via private LDS tile
    const _Float16* pr2 = &pbuf[wave][n16 * 32];
    v16h aprob = cat8(ld8(pr2 + hi * 16), ld8(pr2 + hi * 16 + 8));
#pragma unroll
    for (int t = 0; t < 4; ++t) {
      const _Float16* vr = vt + vplane + (size_t)(16 * t + n16) * TT + j0 + hi * 16;
      acc[t] = wmma32(aprob, ld16(vr), acc[t]);
    }
  }

#pragma unroll
  for (int v = 0; v < 8; ++v) {
    int m = v + 8 * hi;
    float inv = 1.f / lrow[v];
    size_t rb = ((size_t)b * TT + i0 + m) * CC + head * DD;
#pragma unroll
    for (int t = 0; t < 4; ++t)
      yh[rb + 16 * t + n16] = (_Float16)(acc[t][v] * inv);
  }
}

// ------------------------------------------------------- output GEMM (f32)
__global__ __launch_bounds__(256) void fox_out_kernel(
    const _Float16* __restrict__ yh, const _Float16* __restrict__ wpr,
    float* __restrict__ out) {
  const int lane = threadIdx.x & 31;
  const int wave = threadIdx.x >> 5;
  const int n16  = lane & 15;
  const int hi   = lane >> 4;
  const int nbase = blockIdx.x * 64;
  const int i0    = (blockIdx.y * 8 + wave) * 16;

  v8f acc[4];
#pragma unroll
  for (int t = 0; t < 4; ++t) acc[t] = v8f{};

  const _Float16* arow = yh + (size_t)(i0 + n16) * CC;
  for (int kk = 0; kk < CC; kk += 32) {
    __builtin_prefetch(arow + kk + 128, 0, 1);
    v16h a = cat8(ld8(arow + kk + hi * 16), ld8(arow + kk + hi * 16 + 8));
#pragma unroll
    for (int t = 0; t < 4; ++t) {
      v16h bfr = ld16(wpr + (size_t)(nbase + 16 * t + n16) * CC + kk + hi * 16);
      acc[t] = wmma32(a, bfr, acc[t]);
    }
  }
#pragma unroll
  for (int v = 0; v < 8; ++v) {
    int it = i0 + v + 8 * hi;
#pragma unroll
    for (int t = 0; t < 4; ++t)
      out[(size_t)it * CC + nbase + 16 * t + n16] = acc[t][v];
  }
}

// ------------------------------------------------------------------ launch
extern "C" void kernel_launch(void* const* d_in, const int* in_sizes, int n_in,
                              void* d_out, int out_size, void* d_ws,
                              size_t ws_size, hipStream_t stream) {
  (void)in_sizes; (void)n_in; (void)out_size; (void)ws_size;
  const float* x   = (const float*)d_in[0];
  const float* Wq  = (const float*)d_in[1];
  const float* Wk  = (const float*)d_in[2];
  const float* Wv  = (const float*)d_in[3];
  const float* Wp  = (const float*)d_in[4];
  const float* Wpr = (const float*)d_in[5];
  const float* qw  = (const float*)d_in[6];
  const float* kw  = (const float*)d_in[7];

  char* ws = (char*)d_ws;
  const size_t MB = 1ull << 20;
  _Float16* xh    = (_Float16*)(ws + 0 * MB);   // 8 MB
  _Float16* whq   = (_Float16*)(ws + 8 * MB);   // 2 MB each
  _Float16* whk   = (_Float16*)(ws + 10 * MB);
  _Float16* whv   = (_Float16*)(ws + 12 * MB);
  _Float16* whp   = (_Float16*)(ws + 14 * MB);
  _Float16* whpr  = (_Float16*)(ws + 16 * MB);
  _Float16* qh    = (_Float16*)(ws + 18 * MB);  // 8 MB each, [B,H,T,D]
  _Float16* kh    = (_Float16*)(ws + 26 * MB);
  _Float16* ph    = (_Float16*)(ws + 34 * MB);
  _Float16* prh   = (_Float16*)(ws + 42 * MB);
  _Float16* vt    = (_Float16*)(ws + 50 * MB);  // [B,H,D,T]
  _Float16* yh    = (_Float16*)(ws + 58 * MB);  // 8 MB

  fox_f32_to_f16<<<4096, 256, 0, stream>>>(x, xh, 2 * TT * CC);
  fox_f32_to_f16<<<1024, 256, 0, stream>>>(Wq, whq, CC * CC);
  fox_f32_to_f16<<<1024, 256, 0, stream>>>(Wk, whk, CC * CC);
  fox_f32_to_f16<<<1024, 256, 0, stream>>>(Wv, whv, CC * CC);
  fox_f32_to_f16<<<1024, 256, 0, stream>>>(Wp, whp, CC * CC);
  fox_f32_to_f16<<<1024, 256, 0, stream>>>(Wpr, whpr, CC * CC);

  fox_proj_kernel<<<dim3(HH, 32, 4), 256, 0, stream>>>(
      xh, whq, whk, whv, whp, qw, kw, qh, kh, vt, ph, prh);

  fox_attn_kernel<<<dim3(32, HH, 2), 128, 0, stream>>>(qh, kh, ph, prh, vt, yh);

  fox_out_kernel<<<dim3(HH, 32, 1), 256, 0, stream>>>(yh, whpr, (float*)d_out);
}
